// GatedConvNetwork_24653112279616
// MI455X (gfx1250) — matrix operand order repaired and checked by reading, not compiled, not run on hardware
//
#include <hip/hip_runtime.h>
#include <hip/hip_bf16.h>
#include <math.h>

typedef __bf16 bf16_t;
typedef __attribute__((ext_vector_type(16))) bf16_t v16bf;
typedef __attribute__((ext_vector_type(8)))  float  v8f;

#define NATOM 192
#define MULC  16
#define NPAIR (NATOM*NATOM)      // 36864 pairs
#define HDIM  100
#define KPAD  128                // K padded for 16x16x32 bf16 WMMA
#define NTILE_M (NPAIR/16)       // 2304 row tiles

__device__ __forceinline__ float fast_sig(float x){
    return __builtin_amdgcn_rcpf(1.f + __expf(-x));   // v_rcp_f32, no IEEE div chain
}
__device__ __forceinline__ float swishf(float x){ return x * fast_sig(x); }

union ABU { v16bf v; uint4 q[2]; };

// A-operand fragment: lane M = lane&15, khalf = lane>>4, two 16B chunks
__device__ __forceinline__ v16bf load_afrag(const bf16_t* __restrict__ arow, int ks, int kh){
    ABU a;
    int ka = ks*32 + kh*8;
    a.q[0] = *(const uint4*)(arow + ka);
    a.q[1] = *(const uint4*)(arow + ka + 16);
    return a.v;
}
// B-operand fragment: lane N = lane&15, khalf = lane>>4, two 16B chunks
__device__ __forceinline__ v16bf load_bfrag(const bf16_t* __restrict__ brow, int ks, int kh){
    ABU b;
    int kb = ks*32 + kh*8;
    b.q[0] = *(const uint4*)(brow + kb);
    b.q[1] = *(const uint4*)(brow + kb + 16);
    return b.v;
}

// ---------------------------------------------------------------- geometry
__global__ void geom_kernel(const float* __restrict__ pos,
                            float* __restrict__ basis, float* __restrict__ Y1)
{
    int P = blockIdx.x*256 + threadIdx.x;
    if (P >= NPAIR) return;
    int a = P / NATOM, b = P % NATOM;
    float rx = pos[a*3+0]-pos[b*3+0];
    float ry = pos[a*3+1]-pos[b*3+1];
    float rz = pos[a*3+2]-pos[b*3+2];
    float d  = sqrtf(rx*rx+ry*ry+rz*rz + 1e-12f);
    float s3inv = 1.7320508f * __builtin_amdgcn_rcpf(d + 1e-9f);   // sqrt(3)/d
    Y1[P*3+0]=rx*s3inv; Y1[P*3+1]=ry*s3inv; Y1[P*3+2]=rz*s3inv;
#pragma unroll
    for (int j=0;j<3;++j){                          // mu={0,0.5,1}, step=0.5
        float t = (d - 0.5f*j) * 2.0f;
        basis[P*3+j] = __expf(-t*t);
    }
}

// -------------------------------------------- basis @ w0 (K=3) + swish -> bf16
__global__ void h1_kernel(const float* __restrict__ basis, const float* __restrict__ w0,
                          bf16_t* __restrict__ h1)
{
    int idx = blockIdx.x*256 + threadIdx.x;
    if (idx >= NPAIR*KPAD) return;
    int P = idx >> 7, k = idx & 127;
    float v = 0.f;
    if (k < HDIM){
        float s = basis[P*3+0]*w0[0*HDIM+k] + basis[P*3+1]*w0[1*HDIM+k]
                + basis[P*3+2]*w0[2*HDIM+k];
        v = swishf(s * 0.57735027f);                // / sqrt(NB)
    }
    h1[idx] = (bf16_t)v;
}

// -------- transpose+pad weight [HDIM,cols] -> bf16 [colspad,KPAD], scaled
__global__ void wconv_kernel(const float* __restrict__ w, bf16_t* __restrict__ wb,
                             int cols, int colspad, float scale)
{
    int idx = blockIdx.x*256 + threadIdx.x;
    if (idx >= colspad*KPAD) return;
    int c = idx >> 7, k = idx & 127;
    float v = (c < cols && k < HDIM) ? w[k*cols + c]*scale : 0.f;
    wb[idx] = (bf16_t)v;
}

// ----------- h1[36864,128] @ w1b -> swish -> bf16 h2; A in registers for all 7 tiles
__global__ void __launch_bounds__(32)
gemm_swish_kernel(const bf16_t* __restrict__ A, const bf16_t* __restrict__ B,
                  bf16_t* __restrict__ C)
{
    int tm = blockIdx.x;
    int lane = threadIdx.x;
    int m = lane & 15, kh = lane >> 4;
    const bf16_t* arow = A + (size_t)(tm*16 + m)*KPAD;

    v16bf afrag[4];
#pragma unroll
    for (int ks=0; ks<4; ++ks) afrag[ks] = load_afrag(arow, ks, kh);

#pragma unroll
    for (int tn=0; tn<7; ++tn){
        const bf16_t* brow = B + (size_t)(tn*16 + m)*KPAD;
        v8f acc = {};
#pragma unroll
        for (int ks=0; ks<4; ++ks){
            v16bf bfrag = load_bfrag(brow, ks, kh);
            acc = __builtin_amdgcn_wmma_f32_16x16x32_bf16(false, afrag[ks], false, bfrag,
                                                          (short)0, acc, false, false);
        }
#pragma unroll
        for (int r=0;r<8;++r){                      // D: M = r + kh*8, N = m
            int gr = tm*16 + r + kh*8;
            int gc = tn*16 + m;
            C[(size_t)gr*KPAD + gc] = (bf16_t)swishf(acc[r]);
        }
    }
    // zero the pad columns 112..127
#pragma unroll
    for (int r=0;r<8;++r){
        int gr = tm*16 + r + kh*8;
        C[(size_t)gr*KPAD + 112 + m] = (bf16_t)0.f;
    }
}

// ---- fused h2 @ w2 -> R tiles (registers only) -> CG contraction -> atomics
// All 16 rows of a pair tile share the same output atom a, and the CG factors are
// linear in the per-row dots, so they fold into chunk-invariant per-row weights:
// per chunk, p = sum_r acc[r]*w[r], one wave-wide reduce, 1 (or 3) atomics.
__global__ void __launch_bounds__(32)
conv_kernel(const bf16_t* __restrict__ A, const bf16_t* __restrict__ B,
            const float* __restrict__ Y1, const float* __restrict__ x0,
            const float* __restrict__ x1,
            float* __restrict__ sacc, float* __restrict__ gacc, float* __restrict__ vacc,
            int layer)
{
    int tm   = blockIdx.x;                          // 16-pair tile (atom a = tm/12)
    int path = blockIdx.y >> 1;
    int ug   = (blockIdx.y & 1) * 8;                // u in [ug, ug+8)
    int lane = threadIdx.x;
    int m = lane & 15, kh = lane >> 4;

    const float NORM = 0.072168784f;   // 1/sqrt(192)
    const float ISQ3 = 0.57735027f;
    const float ISQ2 = 0.70710678f;

    // weight kind: 0=xa  1=dot(y,xv)  2=xa*y[i]  3=xv[i]  4=cross(y,xv)[i]
    int wkind; float scale; float* dst;
    if (layer == 0){
        if (path == 0)      { wkind=0; dst=sacc; scale=NORM; }
        else if (path == 1) { wkind=0; dst=gacc; scale=NORM; }
        else                { wkind=2; dst=vacc; scale=ISQ3*NORM; }
    } else if (layer == 3){
        if (path == 0)      { wkind=0; dst=sacc; scale=NORM; }
        else                { wkind=1; dst=sacc; scale=ISQ3*NORM; }
    } else {
        switch (path){
        case 0:  wkind=0; dst=sacc; scale=NORM;       break;
        case 1:  wkind=1; dst=sacc; scale=ISQ3*NORM;  break;
        case 2:  wkind=0; dst=gacc; scale=NORM;       break;
        case 3:  wkind=1; dst=gacc; scale=ISQ3*NORM;  break;
        case 4:  wkind=2; dst=vacc; scale=ISQ3*NORM;  break;
        case 5:  wkind=3; dst=vacc; scale=NORM;       break;
        default: wkind=4; dst=vacc; scale=ISQ2*NORM;  break;
        }
    }
    const bool isVec  = (wkind >= 2);
    const bool needX0 = (wkind == 0 || wkind == 2);
    const bool needY  = (wkind == 1 || wkind == 2 || wkind == 4);

    const bf16_t* arow = A + (size_t)(tm*16 + m)*KPAD;
    v16bf afrag[4];
#pragma unroll
    for (int ks=0; ks<4; ++ks) afrag[ks] = load_afrag(arow, ks, kh);

    // hoist per-row, per-lane weights (scale folded in); invariant over the 8 chunks
    float wS[8], wV[8][3];
#pragma unroll
    for (int r=0;r<8;++r){
        int P = tm*16 + kh*8 + r;                   // pair = a*192+b
        int b = P % NATOM;
        float xa = needX0 ? x0[b*MULC + m] : 0.f;
        float xv0=0.f, xv1=0.f, xv2=0.f;
        if (!needX0){
            const float* xp = x1 + (size_t)(b*MULC + m)*3;
            xv0=xp[0]; xv1=xp[1]; xv2=xp[2];
        }
        float y0=0.f, y1=0.f, y2=0.f;
        if (needY){ y0=Y1[P*3+0]; y1=Y1[P*3+1]; y2=Y1[P*3+2]; }
        switch (wkind){
        case 0: wS[r] = xa*scale; break;
        case 1: wS[r] = (y0*xv0 + y1*xv1 + y2*xv2)*scale; break;
        case 2: wV[r][0]=xa*y0*scale; wV[r][1]=xa*y1*scale; wV[r][2]=xa*y2*scale; break;
        case 3: wV[r][0]=xv0*scale;   wV[r][1]=xv1*scale;   wV[r][2]=xv2*scale;   break;
        default:
            wV[r][0]=(y1*xv2 - y2*xv1)*scale;
            wV[r][1]=(y2*xv0 - y0*xv2)*scale;
            wV[r][2]=(y0*xv1 - y1*xv0)*scale;
            break;
        }
    }

    int obase = (tm/12)*MULC + ug;                  // a*MULC + ug

#pragma unroll
    for (int c=0;c<8;++c){
        const bf16_t* brow = B + (size_t)((path*16 + ug + c)*16 + m)*KPAD;
        v8f acc = {};
#pragma unroll
        for (int ks=0;ks<4;++ks){
            v16bf bfrag = load_bfrag(brow, ks, kh);
            acc = __builtin_amdgcn_wmma_f32_16x16x32_bf16(false, afrag[ks], false, bfrag,
                                                          (short)0, acc, false, false);
        }
        if (!isVec){
            float p = 0.f;
#pragma unroll
            for (int r=0;r<8;++r) p = fmaf(acc[r], wS[r], p);
#pragma unroll
            for (int off=1; off<32; off<<=1) p += __shfl_xor(p, off);
            if (lane == 0) atomicAdd(&dst[obase + c], p);
        } else {
            float p0=0.f, p1=0.f, p2=0.f;
#pragma unroll
            for (int r=0;r<8;++r){
                p0 = fmaf(acc[r], wV[r][0], p0);
                p1 = fmaf(acc[r], wV[r][1], p1);
                p2 = fmaf(acc[r], wV[r][2], p2);
            }
#pragma unroll
            for (int off=1; off<32; off<<=1){
                p0 += __shfl_xor(p0, off);
                p1 += __shfl_xor(p1, off);
                p2 += __shfl_xor(p2, off);
            }
            if (lane == 0){
                int o = obase + c;
                atomicAdd(&dst[o*3+0], p0);
                atomicAdd(&dst[o*3+1], p1);
                atomicAdd(&dst[o*3+2], p2);
            }
        }
    }
}

// ----------------------------------------------------- gating nonlinearity
__global__ void gate_kernel(const float* __restrict__ sacc, const float* __restrict__ gacc,
                            const float* __restrict__ vacc,
                            float* __restrict__ x0, float* __restrict__ x1)
{
    int idx = blockIdx.x*256 + threadIdx.x;
    if (idx >= NATOM*MULC) return;
    x0[idx] = swishf(sacc[idx]);
    float g = fast_sig(gacc[idx]);
    x1[idx*3+0] = g*vacc[idx*3+0];
    x1[idx*3+1] = g*vacc[idx*3+1];
    x1[idx*3+2] = g*vacc[idx*3+2];
}

__global__ void fill0_kernel(float* __restrict__ p, int n){
    int i = blockIdx.x*256 + threadIdx.x;
    if (i < n) p[i] = 0.f;
}

// ------------------------------------------------------------------ driver
extern "C" void kernel_launch(void* const* d_in, const int* in_sizes, int n_in,
                              void* d_out, int out_size, void* d_ws, size_t ws_size,
                              hipStream_t stream)
{
    (void)in_sizes; (void)n_in; (void)out_size; (void)ws_size;
    const float* x_in = (const float*)d_in[0];      // [1,192,16]
    const float* pos  = (const float*)d_in[1];      // [1,192,3]
    const float* W[4][3];
    for (int l=0;l<4;++l) for (int j=0;j<3;++j) W[l][j] = (const float*)d_in[2 + l*3 + j];
    const int npaths[4] = {3,7,7,2};

    char* ws = (char*)d_ws;
    size_t off = 0;
    auto alloc = [&](size_t bytes)->char*{
        char* p = ws + off; off = (off + bytes + 255) & ~(size_t)255; return p;
    };
    float*  basis = (float*) alloc((size_t)NPAIR*3*sizeof(float));
    float*  Y1    = (float*) alloc((size_t)NPAIR*3*sizeof(float));
    bf16_t* h1    = (bf16_t*)alloc((size_t)NPAIR*KPAD*2);
    bf16_t* h2    = (bf16_t*)alloc((size_t)NPAIR*KPAD*2);
    bf16_t* w1b   = (bf16_t*)alloc((size_t)112*KPAD*2);
    bf16_t* w2b   = (bf16_t*)alloc((size_t)1792*KPAD*2);
    float*  x0b   = (float*) alloc((size_t)NATOM*MULC*4);
    float*  x1b   = (float*) alloc((size_t)NATOM*MULC*3*4);
    float*  sacc  = (float*) alloc((size_t)NATOM*MULC*4);
    float*  gacc  = (float*) alloc((size_t)NATOM*MULC*4);
    float*  vacc  = (float*) alloc((size_t)NATOM*MULC*3*4);

    geom_kernel<<<(NPAIR+255)/256, 256, 0, stream>>>(pos, basis, Y1);

    const float* x0cur = x_in;
    const float* x1cur = x1b;                       // unread at layer 0
    for (int l=0;l<4;++l){
        int p = npaths[l];
        h1_kernel<<<(NPAIR*KPAD+255)/256, 256, 0, stream>>>(basis, W[l][0], h1);
        wconv_kernel<<<(112*KPAD+255)/256, 256, 0, stream>>>(W[l][1], w1b, HDIM, 112, 0.1f);
        gemm_swish_kernel<<<NTILE_M, 32, 0, stream>>>(h1, w1b, h2);
        int cols = p*256;
        wconv_kernel<<<(cols*KPAD+255)/256, 256, 0, stream>>>(W[l][2], w2b, cols, cols, 0.1f);

        float* sdst = (l==3) ? (float*)d_out : sacc;
        fill0_kernel<<<(NATOM*MULC+255)/256, 256, 0, stream>>>(sdst, NATOM*MULC);
        if (l < 3){
            fill0_kernel<<<(NATOM*MULC+255)/256, 256, 0, stream>>>(gacc, NATOM*MULC);
            fill0_kernel<<<(NATOM*MULC*3+255)/256, 256, 0, stream>>>(vacc, NATOM*MULC*3);
        }
        conv_kernel<<<dim3(NTILE_M, 2*p), 32, 0, stream>>>(h2, w2b, Y1, x0cur, x1cur,
                                                           sdst, gacc, vacc, l);
        if (l < 3){
            gate_kernel<<<(NATOM*MULC+255)/256, 256, 0, stream>>>(sdst, gacc, vacc, x0b, x1b);
            x0cur = x0b; x1cur = x1b;
        }
    }
}
